// FasterRCNN_69329362092687
// MI455X (gfx1250) — matrix operand order repaired and checked by reading, not compiled, not run on hardware
//
#include <hip/hip_runtime.h>

typedef __attribute__((ext_vector_type(16))) __bf16 v16bf;
typedef __attribute__((ext_vector_type(8)))  __bf16 v8bf;
typedef __attribute__((ext_vector_type(8)))  float  v8f;

// -------------------------------------------------------------------------
// Weight pre-pack: HWIO f32 -> bf16 B-fragment layout.
// Packed element i = tile*512 + lane*16 + h holds W[p][cc*32 + K][nt*16 + n]
// with K = 16*(h/8) + 8*(lane/16) + h%8, n = lane%16  (CDNA5 16-bit B-matrix layout).
// -------------------------------------------------------------------------
__global__ void pack_w_k(const float* __restrict__ W, __bf16* __restrict__ outp,
                         int Cin, int Cout, int P) {
  const long i = (long)blockIdx.x * 256 + threadIdx.x;
  const int cchunks = Cin >> 5;
  const int ntiles  = Cout >> 4;
  const long total = (long)P * cchunks * ntiles * 512;
  if (i >= total) return;
  const int h    = (int)(i & 15);
  const int lane = (int)((i >> 4) & 31);
  long tile = i >> 9;
  const int nt = (int)(tile % ntiles); tile /= ntiles;
  const int cc = (int)(tile % cchunks);
  const int p  = (int)(tile / cchunks);
  const int K  = ((h >> 3) << 4) + ((lane >> 4) << 3) + (h & 7);
  const int n  = lane & 15;
  const long src = ((long)p * Cin + cc * 32 + K) * Cout + nt * 16 + n;
  outp[i] = (__bf16)W[src];
}

// -------------------------------------------------------------------------
// First conv: 3x3, Cin=3 (too skinny for K=32 WMMA) -> direct f32, bf16 out.
// -------------------------------------------------------------------------
__global__ void conv0_k(const float* __restrict__ img, const float* __restrict__ W,
                        const float* __restrict__ bias, __bf16* __restrict__ out) {
  const int i  = blockIdx.x * 256 + threadIdx.x;   // pix*64 + co, exact grid
  const int co = i & 63;
  const int pix = i >> 6;
  const int y = pix >> 9, x = pix & 511;
  float s = bias[co];
  for (int ky = 0; ky < 3; ++ky)
    for (int kx = 0; kx < 3; ++kx) {
      const int iy = y + ky - 1, ix = x + kx - 1;
      if (iy < 0 || iy >= 512 || ix < 0 || ix >= 512) continue;
      const float* ip = img + ((iy << 9) + ix) * 3;
      const float* wq = W + (ky * 3 + kx) * 3 * 64 + co;
      s += ip[0] * wq[0] + ip[1] * wq[64] + ip[2] * wq[128];
    }
  s = s > 0.f ? s : 0.f;
  out[i] = (__bf16)s;
}

// -------------------------------------------------------------------------
// Implicit-GEMM conv, bf16 in/out, f32 accumulate, v_wmma_f32_16x16x32_bf16.
//   Templated on <P, CIN> so all geometry is compile-time.
//   block = 256 threads (8 waves). Block owns one 16-pixel M-tile (always a
//   single image row since W % 16 == 0).
//   Stage ONCE per block: full 3x18 (or 1x16) halo x CIN into LDS with b128
//   loads (padded rows for bank skew), one barrier, then 9 taps read LDS only.
//   Each wave computes 16 px x 32 channels: two independent accumulators ->
//   2 WMMAs per A fragment. Inactive waves stage + barrier; compute guarded
//   wave-uniformly so EXEC is all-ones for WMMA.
// -------------------------------------------------------------------------
template <int P, int CIN>
__global__ void conv_wmma_k(const __bf16* __restrict__ in, const __bf16* __restrict__ wpk,
                            const float* __restrict__ bias, __bf16* __restrict__ out,
                            int H, int Wshift, int Cout) {
  constexpr int R       = (P == 9) ? 3 : 1;     // halo rows
  constexpr int C       = (P == 9) ? 18 : 16;   // halo cols
  constexpr int rowH    = CIN + 8;              // padded LDS row (halves)
  constexpr int cchunks = CIN / 32;
  constexpr int units   = CIN / 8;              // b128 units per halo cell
  constexpr int totalUnits = R * C * units;

  __shared__ __bf16 lds[R * C * rowH];

  const int Wd   = 1 << Wshift;
  const int tid  = threadIdx.x;
  const int wave = tid >> 5;
  const int lane = tid & 31;
  const int ntiles = Cout >> 4;
  const int pairs  = Cout >> 5;
  const int pr     = blockIdx.y * 8 + wave;
  const bool active = pr < pairs;               // wave-uniform
  const int nt0 = pr * 2;
  const int mt   = blockIdx.x;
  const int m    = lane & 15;
  const int kgrp = lane >> 4;

  const int pixBase = mt * 16;                  // tile start (single row)
  const int y0 = pixBase >> Wshift;
  const int x0 = pixBase & (Wd - 1);

  // ---- stage halo once (b128, zero-fill SAME padding) ----
  for (int w = tid; w < totalUnits; w += 256) {
    const int cell = w / units;                 // shift (units is pow2)
    const int u    = w - cell * units;
    const int r    = cell / C;
    const int c    = cell - r * C;
    const int iy = y0 + r - ((P == 9) ? 1 : 0);
    const int ix = x0 + c - ((P == 9) ? 1 : 0);
    uint4 v = {0u, 0u, 0u, 0u};
    if (iy >= 0 && iy < H && ix >= 0 && ix < Wd)
      v = *(const uint4*)(in + ((((long)iy << Wshift) + ix) * CIN) + u * 8);
    *(uint4*)(lds + cell * rowH + u * 8) = v;
  }
  __syncthreads();

  v8f acc0 = {}, acc1 = {};

  if (active) {
    for (int p = 0; p < P; ++p) {
      const int dy = (P == 9) ? (p / 3) : 0;
      const int dx = (P == 9) ? (p % 3) : 0;
      const __bf16* afrag = lds + ((dy * C + dx + m) * rowH) + kgrp * 8;
      const __bf16* bptr  = wpk + ((long)(p * cchunks) * ntiles + nt0) * 512 + lane * 16;
#pragma unroll 4
      for (int cc = 0; cc < cchunks; ++cc) {
        const v8bf lo = *(const v8bf*)(afrag + cc * 32);
        const v8bf hi = *(const v8bf*)(afrag + cc * 32 + 16);
        v16bf a;
#pragma unroll
        for (int h2 = 0; h2 < 8; ++h2) { a[h2] = lo[h2]; a[8 + h2] = hi[h2]; }
        const __bf16* bt = bptr + (long)cc * ntiles * 512;
        const v16bf b0 = *(const v16bf*)(bt);
        const v16bf b1 = *(const v16bf*)(bt + 512);
        acc0 = __builtin_amdgcn_wmma_f32_16x16x32_bf16(false, a, false, b0,
                                                       (short)0, acc0, false, false);
        acc1 = __builtin_amdgcn_wmma_f32_16x16x32_bf16(false, a, false, b1,
                                                       (short)0, acc1, false, false);
      }
    }

    // C/D layout: lane l, VGPR r -> M = r + 8*(l/16), N = l%16
    const int n = lane & 15;
    const float bv0 = bias[nt0 * 16 + n];
    const float bv1 = bias[nt0 * 16 + 16 + n];
#pragma unroll
    for (int r = 0; r < 8; ++r) {
      const long orow = (long)pixBase + r + 8 * kgrp;
      float v0 = acc0[r] + bv0;
      float v1 = acc1[r] + bv1;
      v0 = v0 > 0.f ? v0 : 0.f;
      v1 = v1 > 0.f ? v1 : 0.f;
      out[orow * Cout + nt0 * 16 + n]      = (__bf16)v0;
      out[orow * Cout + nt0 * 16 + 16 + n] = (__bf16)v1;
    }
  }
}

// -------------------------------------------------------------------------
// 2x2 stride-2 max pool (bf16 NHWC)
// -------------------------------------------------------------------------
__global__ void pool_k(const __bf16* __restrict__ in, __bf16* __restrict__ out,
                       int Ho, int Wo, int C) {
  const long i = (long)blockIdx.x * 256 + threadIdx.x;
  const long total = (long)Ho * Wo * C;
  if (i >= total) return;
  const int c = (int)(i % C);
  long t = i / C;
  const int xo = (int)(t % Wo);
  const int yo = (int)(t / Wo);
  const int Wi = Wo * 2;
  float mx = -3.4e38f;
  for (int dy = 0; dy < 2; ++dy)
    for (int dx = 0; dx < 2; ++dx) {
      const float v = (float)in[((long)(yo * 2 + dy) * Wi + (xo * 2 + dx)) * C + c];
      mx = v > mx ? v : mx;
    }
  out[i] = (__bf16)mx;
}

// -------------------------------------------------------------------------
// RPN heads (tiny 1x1 convs, f32 math on bf16 feats)
// -------------------------------------------------------------------------
__global__ void cls_k(const __bf16* __restrict__ feats, const float* __restrict__ W,
                      const float* __restrict__ b, float* __restrict__ out) {
  const int i = blockIdx.x * 256 + threadIdx.x;
  if (i >= 1024 * 9) return;
  const int a = i % 9, pix = i / 9;
  const __bf16* f = feats + pix * 256;
  float s = b[a];
  for (int c = 0; c < 256; ++c) s += (float)f[c] * W[c * 9 + a];
  out[i] = 1.f / (1.f + __expf(-s));
}

__global__ void reg_k(const __bf16* __restrict__ feats, const float* __restrict__ W,
                      const float* __restrict__ b, float* __restrict__ out) {
  const int i = blockIdx.x * 256 + threadIdx.x;
  if (i >= 1024 * 36) return;
  const int o = i % 36, pix = i / 36;
  const __bf16* f = feats + pix * 256;
  float s = b[o];
  for (int c = 0; c < 256; ++c) s += (float)f[c] * W[c * 36 + o];
  out[i] = s;
}

// -------------------------------------------------------------------------
// Box decode + validity masking
// -------------------------------------------------------------------------
__global__ void prop_k(const float* __restrict__ cls, const float* __restrict__ reg,
                       const float* __restrict__ ancs, const float* __restrict__ avalid,
                       float* __restrict__ scores, float* __restrict__ boxes) {
  const int i = blockIdx.x * 256 + threadIdx.x;
  if (i >= 9216) return;
  float s = cls[i];
  s = (avalid[i] > 0.f) ? s : -1.f;
  const float* an = ancs + i * 4;
  const float* of = reg + i * 4;
  const float c0 = an[0] + of[0] * an[2];
  const float c1 = an[1] + of[1] * an[3];
  const float w0 = an[2] * __expf(of[2]);
  const float w1 = an[3] * __expf(of[3]);
  boxes[i * 4 + 0] = c0 - w0 * 0.5f;
  boxes[i * 4 + 1] = c1 - w1 * 0.5f;
  boxes[i * 4 + 2] = c0 + w0 * 0.5f;
  boxes[i * 4 + 3] = c1 + w1 * 0.5f;
  scores[i] = s;
}

// -------------------------------------------------------------------------
// Greedy NMS: 2000 iterations of (block argmax -> suppress). Single block.
// -------------------------------------------------------------------------
__global__ void nms_k(float* __restrict__ scores, const float* __restrict__ boxes,
                      float* __restrict__ out) {
  __shared__ float sv[256];
  __shared__ int   si[256];
  const int tid = threadIdx.x;
  const float NEG = -__builtin_inff();
  for (int it = 0; it < 2000; ++it) {
    float bv = NEG; int bi = 0x7fffffff;
    for (int j = tid; j < 9216; j += 256) {
      const float v = scores[j];
      if (v > bv || (v == bv && j < bi)) { bv = v; bi = j; }
    }
    sv[tid] = bv; si[tid] = bi;
    __syncthreads();
    for (int s = 128; s > 0; s >>= 1) {
      if (tid < s) {
        const float ov = sv[tid + s]; const int oi = si[tid + s];
        if (ov > sv[tid] || (ov == sv[tid] && oi < si[tid])) { sv[tid] = ov; si[tid] = oi; }
      }
      __syncthreads();
    }
    const int   idx  = si[0];
    const float smax = sv[0];
    __syncthreads();
    const float b0 = boxes[idx * 4 + 0], b1 = boxes[idx * 4 + 1];
    const float b2 = boxes[idx * 4 + 2], b3 = boxes[idx * 4 + 3];
    const float barea = (b2 - b0) * (b3 - b1);
    for (int j = tid; j < 9216; j += 256) {
      const float a0 = boxes[j * 4 + 0], a1 = boxes[j * 4 + 1];
      const float a2 = boxes[j * 4 + 2], a3 = boxes[j * 4 + 3];
      const float hh = fmaxf(fminf(b2, a2) - fmaxf(b0, a0), 0.f);
      const float ww = fmaxf(fminf(b3, a3) - fmaxf(b1, a1), 0.f);
      const float inter = hh * ww;
      const float area  = (a2 - a0) * (a3 - a1);
      const float iou   = inter / (area + barea - inter);
      if (iou > 0.7f) scores[j] = NEG;
    }
    if (tid == 0) {
      const bool keep = smax > 0.f;
      out[it * 4 + 0] = keep ? b0 : 0.f;
      out[it * 4 + 1] = keep ? b1 : 0.f;
      out[it * 4 + 2] = keep ? b2 : 0.f;
      out[it * 4 + 3] = keep ? b3 : 0.f;
    }
    __syncthreads();
  }
}

// -------------------------------------------------------------------------
extern "C" void kernel_launch(void* const* d_in, const int* in_sizes, int n_in,
                              void* d_out, int out_size, void* d_ws, size_t ws_size,
                              hipStream_t stream) {
  (void)in_sizes; (void)n_in; (void)out_size; (void)ws_size;
  const float* img    = (const float*)d_in[0];
  const float* ancs   = (const float*)d_in[1];
  const float* avalid = (const float*)d_in[2];
  const float* W[16]; const float* B[16];
  for (int i = 0; i < 16; ++i) { W[i] = (const float*)d_in[3 + 2 * i];
                                 B[i] = (const float*)d_in[4 + 2 * i]; }

  char* ws = (char*)d_ws;
  __bf16* bufA   = (__bf16*)ws;                              // 512*512*64 bf16 = 32 MiB
  __bf16* bufB   = (__bf16*)(ws + (size_t)33554432);
  __bf16* wpBase = (__bf16*)(ws + (size_t)67108864);

  static const int cins [12] = {64, 64, 128, 128, 256, 256, 256, 512, 512, 512, 512, 512};
  static const int couts[12] = {64, 128, 128, 256, 256, 256, 512, 512, 512, 512, 512, 512};
  __bf16* wp[13];
  size_t off = 0;
  for (int l = 0; l < 12; ++l) { wp[l] = wpBase + off; off += (size_t)9 * cins[l] * couts[l]; }
  wp[12] = wpBase + off; off += (size_t)512 * 256;           // bottleneck 1x1
  __bf16* feats = wpBase + off;                              // 32*32*256 bf16
  float* scores = (float*)(feats + 32 * 32 * 256);
  float* boxes  = scores + 9216;

  float* out_cls  = (float*)d_out;
  float* out_reg  = out_cls + 9216;
  float* out_prop = out_reg + 36864;

  // Pack weights (every call; deterministic)
  for (int l = 0; l < 12; ++l) {
    const long tot = (long)9 * cins[l] * couts[l];
    pack_w_k<<<dim3((unsigned)((tot + 255) / 256)), 256, 0, stream>>>(
        W[l + 1], wp[l], cins[l], couts[l], 9);
  }
  pack_w_k<<<dim3((512 * 256 + 255) / 256), 256, 0, stream>>>(W[13], wp[12], 512, 256, 1);

  // conv1 direct (3 -> 64)
  conv0_k<<<dim3(512 * 512 * 64 / 256), 256, 0, stream>>>(img, W[0], B[0], bufA);

  auto conv = [&](const __bf16* in, __bf16* outp, const __bf16* wpk, const float* bias,
                  int H, int Wshift, int Cin, int Cout) {
    const int pairs = Cout / 32;
    dim3 g((unsigned)((H << Wshift) / 16), (unsigned)((pairs + 7) / 8));
    switch (Cin) {
      case 64:  conv_wmma_k<9, 64 ><<<g, 256, 0, stream>>>(in, wpk, bias, outp, H, Wshift, Cout); break;
      case 128: conv_wmma_k<9, 128><<<g, 256, 0, stream>>>(in, wpk, bias, outp, H, Wshift, Cout); break;
      case 256: conv_wmma_k<9, 256><<<g, 256, 0, stream>>>(in, wpk, bias, outp, H, Wshift, Cout); break;
      default:  conv_wmma_k<9, 512><<<g, 256, 0, stream>>>(in, wpk, bias, outp, H, Wshift, Cout); break;
    }
  };
  auto pool = [&](const __bf16* in, __bf16* outp, int Ho, int Wo, int C) {
    const long tot = (long)Ho * Wo * C;
    pool_k<<<dim3((unsigned)((tot + 255) / 256)), 256, 0, stream>>>(in, outp, Ho, Wo, C);
  };

  conv(bufA, bufB, wp[0],  B[1], 512, 9, 64, 64);
  pool(bufB, bufA, 256, 256, 64);
  conv(bufA, bufB, wp[1],  B[2], 256, 8, 64, 128);
  conv(bufB, bufA, wp[2],  B[3], 256, 8, 128, 128);
  pool(bufA, bufB, 128, 128, 128);
  conv(bufB, bufA, wp[3],  B[4], 128, 7, 128, 256);
  conv(bufA, bufB, wp[4],  B[5], 128, 7, 256, 256);
  conv(bufB, bufA, wp[5],  B[6], 128, 7, 256, 256);
  pool(bufA, bufB, 64, 64, 256);
  conv(bufB, bufA, wp[6],  B[7], 64, 6, 256, 512);
  conv(bufA, bufB, wp[7],  B[8], 64, 6, 512, 512);
  conv(bufB, bufA, wp[8],  B[9], 64, 6, 512, 512);
  pool(bufA, bufB, 32, 32, 512);
  conv(bufB, bufA, wp[9],  B[10], 32, 5, 512, 512);
  conv(bufA, bufB, wp[10], B[11], 32, 5, 512, 512);
  conv(bufB, bufA, wp[11], B[12], 32, 5, 512, 512);

  // bottleneck 1x1 512->256 (WMMA, P=1) -> feats
  conv_wmma_k<1, 512><<<dim3((32 * 32) / 16, 1), 256, 0, stream>>>(
      bufA, wp[12], B[13], feats, 32, 5, 256);

  cls_k <<<dim3(36),  256, 0, stream>>>(feats, W[14], B[14], out_cls);
  reg_k <<<dim3(144), 256, 0, stream>>>(feats, W[15], B[15], out_reg);
  prop_k<<<dim3(36),  256, 0, stream>>>(out_cls, out_reg, ancs, avalid, scores, boxes);
  nms_k <<<dim3(1),   256, 0, stream>>>(scores, boxes, out_prop);
}
